// CausalSelfAttention_58815282151616
// MI455X (gfx1250) — compile-verified, matrix operand order checked by reading, not run on hardware
//
#include <hip/hip_runtime.h>
#include <hip/hip_bf16.h>
#include <stdint.h>

// ---------------- problem constants (B=2, T=2048, C=1024, H=16, D=64) ------
#define BB 2
#define TT 2048
#define HH 16
#define DD 64
#define CC 1024
#define N3 3072
#define MR 4096   // B*T

typedef __attribute__((ext_vector_type(16))) __bf16 v16bf;
typedef __attribute__((ext_vector_type(8)))  float  v8f;

struct alignas(16) U4 { unsigned int x, y, z, w; };
union Frag16 { U4 u[2]; v16bf v; };

__device__ __forceinline__ unsigned short f2bf(float f) {
  unsigned int u = __builtin_bit_cast(unsigned int, f);
  u += 0x7FFFu + ((u >> 16) & 1u);          // round-to-nearest-even
  return (unsigned short)(u >> 16);
}

// async global->LDS copy of 16 bytes per lane (gfx1250 async-DMA path,
// tracked by ASYNCcnt). ldsByteOff is the wave-relative LDS byte address.
__device__ __forceinline__ void async_copy_b128(unsigned ldsByteOff,
                                                const unsigned short* g) {
  asm volatile("global_load_async_to_lds_b128 %0, %1, off"
               :: "v"(ldsByteOff), "v"(g) : "memory");
}
__device__ __forceinline__ void wait_asynccnt0() {
  asm volatile("s_wait_asynccnt 0x0" ::: "memory");
}

// ---------------- elementwise f32 -> bf16 ----------------------------------
__global__ void k_f32_to_bf16(const float* __restrict__ in,
                              unsigned short* __restrict__ out, int n) {
  int i = blockIdx.x * 256 + threadIdx.x;
  if (i < n) out[i] = f2bf(in[i]);
}

// ------- pack row-major f32 weight (K x N) into WMMA B-fragment layout -----
// tile = (kt, nt): 32(K) x 16(N). Packed element (lane, i):
//   k_in = 8*(lane>=16) + (i<8 ? i : i+8), n_in = lane&15
__global__ void k_pack_w(const float* __restrict__ W,
                         unsigned short* __restrict__ out, int K, int N) {
  int idx = blockIdx.x * 256 + threadIdx.x;
  if (idx >= K * N) return;
  int tile = idx >> 9, rem = idx & 511;
  int lane = rem >> 4, i = rem & 15;
  int ntiles = N >> 4;
  int kt = tile / ntiles, nt = tile % ntiles;
  int kin = ((lane & 16) ? 8 : 0) + (i < 8 ? i : i + 8);
  int k = kt * 32 + kin;
  int n = nt * 16 + (lane & 15);
  out[idx] = f2bf(W[(size_t)k * N + n]);
}

// ------------- bf16 GEMM: C(f32, MxN) = A(bf16 row-major MxK) @ Bpacked ----
// Block = 8 waves sharing one 64-row A band (block computes 64 x 256 of C).
// A tile (64x32 bf16 = 4KB) is double-buffered in LDS via async global->LDS
// copies; each wave computes a 64(M) x 32(N) tile as 4x2 WMMA 16x16 tiles.
// Requires (N/32) % 8 == 0 and an exact grid (no early-returning waves).
__global__ void __launch_bounds__(256) k_gemm_bf16(
    const unsigned short* __restrict__ A, const unsigned short* __restrict__ Bp,
    float* __restrict__ C, int M, int N, int K) {
  __shared__ unsigned short Atile[2][64 * 32];   // 2 x 4KB double buffer

  int tid  = threadIdx.x;
  int lane = tid & 31;
  int wave = tid >> 5;
  int nWaveN = N >> 5;
  int blocksPerM = nWaveN >> 3;                  // 8 waves per block
  int wm = blockIdx.x / blocksPerM;
  int wn = (blockIdx.x % blocksPerM) * 8 + wave;
  int base  = (lane & 16) ? 8 : 0;               // K (and C-row) sub-offset
  int mlane = lane & 15;

  // async-copy mapping: 256 threads x 16B = 4KB tile; row = tid/4, 16B chunk
  int arow = tid >> 2;                           // 0..63
  int acol = (tid & 3) * 8;                      // element offset 0,8,16,24
  const unsigned short* aGlob0 = A + (size_t)(wm * 64 + arow) * K + acol;
  unsigned ldsOff0 = (unsigned)(size_t)&Atile[0][arow * 32 + acol];

  v8f acc[4][2];
  for (int a = 0; a < 4; ++a)
    for (int b = 0; b < 2; ++b)
      for (int r = 0; r < 8; ++r) acc[a][b][r] = 0.f;

  int ktiles  = K >> 5;
  int ntile16 = N >> 4;

  // prologue: stage kt=0 into buffer 0
  async_copy_b128(ldsOff0, aGlob0);

  for (int kt = 0; kt < ktiles; ++kt) {
    int buf = kt & 1;
    wait_asynccnt0();       // own async copy done
    __syncthreads();        // all copies done; prior buffer reads consumed
    if (kt + 1 < ktiles)
      async_copy_b128(ldsOff0 + (buf ^ 1) * 4096, aGlob0 + (kt + 1) * 32);

    // prefetch next K-step's packed-B tiles (global_prefetch_b8)
    if (kt + 1 < ktiles) {
      const unsigned short* pf =
          Bp + ((size_t)((kt + 1) * ntile16 + wn * 2) << 9) + lane * 16;
      __builtin_prefetch(pf, 0, 1);
      __builtin_prefetch(pf + 512, 0, 1);
    }

    Frag16 af[4];
    for (int mi = 0; mi < 4; ++mi) {
      const unsigned short* ap = &Atile[buf][(mi * 16 + mlane) * 32 + base];
      af[mi].u[0] = ((const U4*)ap)[0];
      af[mi].u[1] = ((const U4*)ap)[2];    // +16 elements = +32B
    }
    Frag16 bfg[2];
    for (int ni = 0; ni < 2; ++ni) {
      const unsigned short* bp =
          Bp + ((size_t)(kt * ntile16 + wn * 2 + ni) << 9) + lane * 16;
      bfg[ni].u[0] = ((const U4*)bp)[0];
      bfg[ni].u[1] = ((const U4*)bp)[1];
    }
    for (int mi = 0; mi < 4; ++mi)
      for (int ni = 0; ni < 2; ++ni)
        acc[mi][ni] = __builtin_amdgcn_wmma_f32_16x16x32_bf16(
            false, af[mi].v, false, bfg[ni].v, (short)0, acc[mi][ni],
            false, false);
  }
  for (int mi = 0; mi < 4; ++mi)
    for (int ni = 0; ni < 2; ++ni)
      for (int r = 0; r < 8; ++r) {
        int row = wm * 64 + mi * 16 + base + r;
        C[(size_t)row * N + wn * 32 + ni * 16 + mlane] = acc[mi][ni][r];
      }
}

// ------------- RoPE + layout repack:  qkv(f32) -> q16/k16 (B,H,T,D), vT ----
__global__ void k_rope_pack(const float* __restrict__ qkv,
                            const float* __restrict__ fc,
                            const float* __restrict__ fs,
                            unsigned short* __restrict__ q16,
                            unsigned short* __restrict__ k16,
                            unsigned short* __restrict__ vt16) {
  int idx = blockIdx.x * 256 + threadIdx.x;  // B*H*T*32 threads exactly
  int d2 = idx & 31;
  int t  = (idx >> 5) & (TT - 1);
  int bh = idx >> 16;                        // 0..31
  int b = bh >> 4, h = bh & 15;
  size_t rb = ((size_t)(b * TT + t)) * N3 + h * DD + 2 * d2;
  float q0 = qkv[rb],          q1 = qkv[rb + 1];
  float k0 = qkv[rb + CC],     k1 = qkv[rb + CC + 1];
  float v0 = qkv[rb + 2 * CC], v1 = qkv[rb + 2 * CC + 1];
  float c = fc[t * 32 + d2], s = fs[t * 32 + d2];
  size_t qo = ((size_t)bh * TT + t) * DD + 2 * d2;
  q16[qo]     = f2bf(q0 * c - q1 * s);
  q16[qo + 1] = f2bf(q0 * s + q1 * c);
  k16[qo]     = f2bf(k0 * c - k1 * s);
  k16[qo + 1] = f2bf(k0 * s + k1 * c);
  size_t vo = ((size_t)bh * DD + 2 * d2) * TT + t;   // transposed (D,T)
  vt16[vo]      = f2bf(v0);
  vt16[vo + TT] = f2bf(v1);
}

// ------------- flash causal attention: one wave per (b,h, 16-row q tile) ---
__global__ void __launch_bounds__(256) k_attn(
    const unsigned short* __restrict__ q16, const unsigned short* __restrict__ k16,
    const unsigned short* __restrict__ vt16, unsigned short* __restrict__ y16) {
  __shared__ unsigned short smem[8][512];    // per-wave 16x32 bf16 P staging
  int lane = threadIdx.x & 31;
  int wave = threadIdx.x >> 5;
  int gw = blockIdx.x * 8 + wave;
  int qb = gw & 127;
  int bh = gw >> 7;
  int b = bh >> 4, h = bh & 15;
  int base = (lane & 16) ? 8 : 0;
  int nl = lane & 15;
  int q0row = qb * 16;

  Frag16 qa[2];
  for (int kc = 0; kc < 2; ++kc) {
    const unsigned short* qp =
        q16 + ((size_t)bh * TT + q0row + nl) * DD + kc * 32 + base;
    qa[kc].u[0] = ((const U4*)qp)[0];
    qa[kc].u[1] = ((const U4*)qp)[2];
  }

  float mrow[8], lrow[8];
  v8f o[4];
  for (int r = 0; r < 8; ++r) { mrow[r] = -__builtin_inff(); lrow[r] = 0.f; }
  for (int nt = 0; nt < 4; ++nt)
    for (int r = 0; r < 8; ++r) o[nt][r] = 0.f;

  const float scale = 0.125f;                // 1/sqrt(64)
  int nj = (q0row + 16 + 31) >> 5;           // 32-key chunks up to causal edge
  unsigned short* sp = smem[wave];

  for (int jb = 0; jb < nj; ++jb) {
    int j0 = jb * 32;
    v8f s0, s1;
    for (int r = 0; r < 8; ++r) { s0[r] = 0.f; s1[r] = 0.f; }
    for (int kc = 0; kc < 2; ++kc) {
      Frag16 kb0, kb1;
      const unsigned short* kp0 =
          k16 + ((size_t)bh * TT + j0 + nl) * DD + kc * 32 + base;
      kb0.u[0] = ((const U4*)kp0)[0]; kb0.u[1] = ((const U4*)kp0)[2];
      const unsigned short* kp1 = kp0 + 16 * DD;
      kb1.u[0] = ((const U4*)kp1)[0]; kb1.u[1] = ((const U4*)kp1)[2];
      s0 = __builtin_amdgcn_wmma_f32_16x16x32_bf16(false, qa[kc].v, false,
                                                   kb0.v, (short)0, s0, false, false);
      s1 = __builtin_amdgcn_wmma_f32_16x16x32_bf16(false, qa[kc].v, false,
                                                   kb1.v, (short)0, s1, false, false);
    }
    // causal mask + online softmax (row reductions across 16 lanes)
    float p0[8], p1[8], corr[8];
    for (int r = 0; r < 8; ++r) {
      int qi = q0row + base + r;
      float a0 = (j0 + nl      <= qi) ? s0[r] * scale : -__builtin_inff();
      float a1 = (j0 + 16 + nl <= qi) ? s1[r] * scale : -__builtin_inff();
      float mx = fmaxf(a0, a1);
      for (int m = 1; m < 16; m <<= 1) mx = fmaxf(mx, __shfl_xor(mx, m, 32));
      float nm = fmaxf(mrow[r], mx);
      float e0 = __expf(a0 - nm);
      float e1 = __expf(a1 - nm);
      float sum = e0 + e1;
      for (int m = 1; m < 16; m <<= 1) sum += __shfl_xor(sum, m, 32);
      corr[r] = __expf(mrow[r] - nm);
      lrow[r] = lrow[r] * corr[r] + sum;
      mrow[r] = nm;
      p0[r] = e0; p1[r] = e1;
    }
    for (int nt = 0; nt < 4; ++nt)
      for (int r = 0; r < 8; ++r) o[nt][r] *= corr[r];

    // transpose P (C-layout -> A-fragment) through LDS
    for (int r = 0; r < 8; ++r) {
      sp[(base + r) * 32 + nl]      = f2bf(p0[r]);
      sp[(base + r) * 32 + 16 + nl] = f2bf(p1[r]);
    }
    asm volatile("s_wait_dscnt 0x0" ::: "memory");  // DS in-order per wave
    Frag16 pa;
    const unsigned short* pp = sp + nl * 32 + base;
    pa.u[0] = ((const U4*)pp)[0];
    pa.u[1] = ((const U4*)pp)[2];

    for (int nt = 0; nt < 4; ++nt) {
      Frag16 vb;
      const unsigned short* vp =
          vt16 + ((size_t)bh * DD + nt * 16 + nl) * TT + j0 + base;
      vb.u[0] = ((const U4*)vp)[0];
      vb.u[1] = ((const U4*)vp)[2];
      o[nt] = __builtin_amdgcn_wmma_f32_16x16x32_bf16(false, pa.v, false,
                                                      vb.v, (short)0, o[nt], false, false);
    }
  }

  for (int r = 0; r < 8; ++r) lrow[r] = 1.f / lrow[r];
  for (int nt = 0; nt < 4; ++nt)
    for (int r = 0; r < 8; ++r) {
      int t = q0row + base + r;
      y16[((size_t)(b * TT + t)) * CC + h * DD + nt * 16 + nl] =
          f2bf(o[nt][r] * lrow[r]);
    }
}

// ---------------------------------------------------------------------------
extern "C" void kernel_launch(void* const* d_in, const int* in_sizes, int n_in,
                              void* d_out, int out_size, void* d_ws, size_t ws_size,
                              hipStream_t stream) {
  (void)in_sizes; (void)n_in; (void)out_size; (void)ws_size;
  const float* x    = (const float*)d_in[0];
  const float* fc   = (const float*)d_in[1];
  const float* fs   = (const float*)d_in[2];
  const float* Wqkv = (const float*)d_in[3];
  const float* Wo   = (const float*)d_in[4];
  float* out = (float*)d_out;

  char* ws = (char*)d_ws;
  size_t off = 0;
  auto take = [&](size_t bytes) -> char* {
    char* p = ws + off;
    off = (off + bytes + 255) & ~(size_t)255;
    return p;
  };
  unsigned short* x16    = (unsigned short*)take((size_t)MR * CC * 2);
  unsigned short* wqkv16 = (unsigned short*)take((size_t)CC * N3 * 2);
  unsigned short* wo16   = (unsigned short*)take((size_t)CC * CC * 2);
  float*          qkv32  = (float*)take((size_t)MR * N3 * 4);
  unsigned short* q16    = (unsigned short*)take((size_t)BB * HH * TT * DD * 2);
  unsigned short* k16    = (unsigned short*)take((size_t)BB * HH * TT * DD * 2);
  unsigned short* vt16   = (unsigned short*)take((size_t)BB * HH * TT * DD * 2);
  unsigned short* y16    = (unsigned short*)take((size_t)MR * CC * 2);

  k_f32_to_bf16<<<(MR * CC) / 256, 256, 0, stream>>>(x, x16, MR * CC);
  k_pack_w<<<(CC * N3) / 256, 256, 0, stream>>>(Wqkv, wqkv16, CC, N3);
  k_pack_w<<<(CC * CC) / 256, 256, 0, stream>>>(Wo, wo16, CC, CC);
  // qkv = x @ Wqkv : 64 row-bands x 12 blocks/band = 768 blocks (exact grid)
  k_gemm_bf16<<<768, 256, 0, stream>>>(x16, wqkv16, qkv32, MR, N3, CC);
  k_rope_pack<<<(BB * HH * TT * 32) / 256, 256, 0, stream>>>(qkv32, fc, fs, q16, k16, vt16);
  // attention: 2*16*128 = 4096 waves -> 512 blocks
  k_attn<<<512, 256, 0, stream>>>(q16, k16, vt16, y16);
  // out = y @ Wo : 64 row-bands x 4 blocks/band = 256 blocks (exact grid)
  k_gemm_bf16<<<256, 256, 0, stream>>>(y16, wo16, out, MR, CC, CC);
}
